// OurModel_80195629351541
// MI455X (gfx1250) — matrix operand order repaired and checked by reading, not compiled, not run on hardware
//
#include <hip/hip_runtime.h>
#include <hip/hip_bf16.h>

// ---------------------------------------------------------------------------
// BiRNN(tanh) + linear + softmax + CRF (loss + viterbi) for gfx1250 (MI455X)
//
// Stage 1 (k_prep)      : pad/convert all weights to bf16 into workspace.
// Stage 2 (k_input_gemm): pre[b,t,:] = x @ W_ih^T + b_ih + b_hh  (both dirs).
//                         bf16 WMMA GEMM, 7 n-tiles fused per wave so each
//                         f32 x tile is loaded+converted once (hw v_cvt).
// Stage 3 (k_rnn)       : sequential scan, h = tanh(pre + h @ W_hh^T).
//                         7 waves/WG: each wave owns a 16-col tile of h,
//                         W_hh fragments resident in VGPRs, h staged in LDS
//                         (b128 ds loads), prefetch of next-step pre rows.
// Stage 4 (k_emis_gemm) : logits = h_cat @ w_lin^T + b_lin (bf16 WMMA).
// Stage 5 (k_softmax)   : emissions = softmax(logits) in place.
// Stage 6 (k_crf)       : per-batch wave32, lane-per-tag forward algorithm
//                         (online logsumexp via shuffles) + gold-path score.
// Stage 7 (k_loss)      : deterministic tree reduction -> d_out[0] = loss.
// Stage 8 (k_viterbi)   : lane-per-tag max-product forward + lane-0 backtrack,
//                         labels (as float, -1 padded) -> d_out[1..].
//
// Workspace requirement: ~203 MB.
// ---------------------------------------------------------------------------

typedef __attribute__((ext_vector_type(16))) __bf16 v16bf;
typedef __attribute__((ext_vector_type(8)))  __bf16 v8bf;
typedef __attribute__((ext_vector_type(8)))  float  v8f;

#define BSZ   256
#define TLEN  512
#define INDIM 202
#define HID   100
#define KTAG  19
#define NTOK  (BSZ * TLEN)      // 131072
#define KPAD  224               // 202 -> 7 * 32
#define HPAD  112               // 100 -> 7 * 16
#define HKP   128               // 100 -> 4 * 32 (recurrent K)
#define ESTR  32                // emission row stride (19 -> 32)
#define NEGBIG (-3.0e38f)

// Hardware f32 -> bf16 conversion (RNE on gfx1250), avoids 3-VALU bit trick.
__device__ __forceinline__ __bf16 f2bf(float f) { return (__bf16)f; }

// ---------------------------------------------------------------------------
// Workspace layout (bytes)
// ---------------------------------------------------------------------------
constexpr size_t SZ_WCAT = (size_t)KPAD * KPAD * 2;        // [224 n][224 k] bf16
constexpr size_t SZ_WHH  = (size_t)HPAD * HKP * 2;         // [112 n][128 k] bf16
constexpr size_t SZ_WLIN = (size_t)32 * KPAD * 2;          // [32 n][224 k] bf16
constexpr size_t OFF_WCAT = 0;
constexpr size_t OFF_WHHF = OFF_WCAT + SZ_WCAT;
constexpr size_t OFF_WHHB = OFF_WHHF + SZ_WHH;
constexpr size_t OFF_WLIN = OFF_WHHB + SZ_WHH;
constexpr size_t OFF_BCAT = OFF_WLIN + SZ_WLIN;            // 224 f32
constexpr size_t OFF_BLIN = OFF_BCAT + 224 * 4;            // 32 f32
constexpr size_t OFF_PREF = ((OFF_BLIN + 32 * 4 + 255) / 256) * 256;
constexpr size_t SZ_PRE   = (size_t)NTOK * HPAD * 4;       // f32 [tok][112]
constexpr size_t OFF_PREB = OFF_PREF + SZ_PRE;
constexpr size_t OFF_HF   = OFF_PREB + SZ_PRE;
constexpr size_t SZ_H     = (size_t)NTOK * HPAD * 2;       // bf16 [tok][112]
constexpr size_t OFF_HB   = OFF_HF + SZ_H;
constexpr size_t OFF_EM   = OFF_HB + SZ_H;
constexpr size_t SZ_EM    = (size_t)NTOK * ESTR * 4;       // f32 [tok][32]
constexpr size_t OFF_HIST = OFF_EM + SZ_EM;
constexpr size_t SZ_HIST  = (size_t)BSZ * (TLEN - 1) * KTAG * 4;   // int
constexpr size_t OFF_LLH  = OFF_HIST + SZ_HIST;            // 256 f32

// ---------------------------------------------------------------------------
// Stage 1: weight prep (pad + f32 -> bf16)
// ---------------------------------------------------------------------------
__global__ void k_prep(const float* __restrict__ w_ih_f, const float* __restrict__ w_hh_f,
                       const float* __restrict__ b_ih_f, const float* __restrict__ b_hh_f,
                       const float* __restrict__ w_ih_b, const float* __restrict__ w_hh_b,
                       const float* __restrict__ b_ih_b, const float* __restrict__ b_hh_b,
                       const float* __restrict__ w_lin,  const float* __restrict__ b_lin,
                       __bf16* Wcat, __bf16* WhhF, __bf16* WhhB, __bf16* Wlin,
                       float* bcat, float* blin) {
    int idx = blockIdx.x * blockDim.x + threadIdx.x;
    const int N_WCAT = KPAD * KPAD, N_WHH = HPAD * HKP, N_WLIN = 32 * KPAD;
    if (idx < N_WCAT) {
        int n = idx / KPAD, k = idx % KPAD; float v = 0.f;
        if (n < HPAD) { if (n < HID && k < INDIM) v = w_ih_f[n * INDIM + k]; }
        else          { int nn = n - HPAD; if (nn < HID && k < INDIM) v = w_ih_b[nn * INDIM + k]; }
        Wcat[idx] = f2bf(v); return;
    }
    idx -= N_WCAT;
    if (idx < N_WHH) {
        int n = idx / HKP, k = idx % HKP;
        WhhF[idx] = f2bf((n < HID && k < HID) ? w_hh_f[n * HID + k] : 0.f); return;
    }
    idx -= N_WHH;
    if (idx < N_WHH) {
        int n = idx / HKP, k = idx % HKP;
        WhhB[idx] = f2bf((n < HID && k < HID) ? w_hh_b[n * HID + k] : 0.f); return;
    }
    idx -= N_WHH;
    if (idx < N_WLIN) {
        int n = idx / KPAD, k = idx % KPAD; float v = 0.f;
        if (n < KTAG) {
            if (k < HID) v = w_lin[n * 2 * HID + k];                     // forward half
            else if (k >= HPAD && k < HPAD + HID) v = w_lin[n * 2 * HID + HID + (k - HPAD)];
        }
        Wlin[idx] = f2bf(v); return;
    }
    idx -= N_WLIN;
    if (idx < KPAD) {
        float v = 0.f;
        if (idx < HID) v = b_ih_f[idx] + b_hh_f[idx];
        else if (idx >= HPAD && idx < HPAD + HID) v = b_ih_b[idx - HPAD] + b_hh_b[idx - HPAD];
        bcat[idx] = v; return;
    }
    idx -= KPAD;
    if (idx < 32) { blin[idx] = (idx < KTAG) ? b_lin[idx] : 0.f; return; }
}

// ---------------------------------------------------------------------------
// Stage 2: pre = x @ Wcat^T + bias   (M = 131072, N = 112/dir, K = 224)
// grid(2048, 2) [m, dir], block 128 (4 waves). Each wave computes a full
// 16 x 112 output strip: the f32 A tile is loaded + converted ONCE per
// k-tile and reused for 7 WMMAs (7 accumulators resident in VGPRs).
// ---------------------------------------------------------------------------
__global__ void k_input_gemm(const float* __restrict__ x, const __bf16* __restrict__ Wcat,
                             const float* __restrict__ bcat,
                             float* __restrict__ pre_f, float* __restrict__ pre_b) {
    const int lane  = threadIdx.x & 31;
    const int wave  = threadIdx.x >> 5;
    const int mtile = blockIdx.x * 4 + wave;
    const int dir   = blockIdx.y;                 // 0 fwd, 1 bwd
    const int half  = lane >> 4, col = lane & 15;
    const int rowTok = mtile * 16 + col;          // A-row for this lane
    const float* xr = x + (size_t)rowTok * INDIM;
    v8f acc[7] = {};
#pragma unroll
    for (int kt = 0; kt < 7; ++kt) {
        v16bf a;
        const int ka = kt * 32 + half * 16;       // A: K in {ka..ka+15}
#pragma unroll
        for (int e = 0; e < 16; ++e) {
            int k = ka + e;
            a[e] = (k < INDIM) ? f2bf(xr[k]) : f2bf(0.f);
        }
        const int kb = kt * 32 + half * 16;       // B: 16 consecutive K (32B vector)
#pragma unroll
        for (int nt = 0; nt < 7; ++nt) {
            const __bf16* wr = Wcat + (size_t)(dir * HPAD + nt * 16 + col) * KPAD;
            v16bf bm = *(const v16bf*)(wr + kb);
            acc[nt] = __builtin_amdgcn_wmma_f32_16x16x32_bf16(false, a, false, bm,
                                                              (short)0, acc[nt], false, false);
        }
    }
    float* dst = dir ? pre_b : pre_f;
#pragma unroll
    for (int nt = 0; nt < 7; ++nt) {
        const float bias = bcat[dir * HPAD + nt * 16 + col];
#pragma unroll
        for (int r = 0; r < 8; ++r) {
            int m = r + half * 8;
            dst[(size_t)(mtile * 16 + m) * HPAD + nt * 16 + col] = acc[nt][r] + bias;
        }
    }
}

// ---------------------------------------------------------------------------
// Stage 3: recurrence h_t = tanh(pre_t + h_{t-1} @ W_hh^T)
// grid(16, 2) [batch-tile, dir], block 224 = 7 waves (one per 16-col tile).
// W_hh fragments live in VGPRs; h round-trips through 4KB LDS per step.
// ---------------------------------------------------------------------------
__global__ void k_rnn(const float* __restrict__ pre_f, const float* __restrict__ pre_b,
                      const __bf16* __restrict__ WhhF, const __bf16* __restrict__ WhhB,
                      __bf16* __restrict__ h_f, __bf16* __restrict__ h_b) {
    __shared__ __bf16 hsh[16][HKP];               // 16 batch rows x 128 K (pad zeroed)
    const int dir   = blockIdx.y;
    const int btile = blockIdx.x;
    const int wave  = threadIdx.x >> 5;           // n-tile 0..6
    const int lane  = threadIdx.x & 31;
    const int half  = lane >> 4, col = lane & 15;
    const int ncol  = wave * 16 + col;
    const __bf16* Whh = dir ? WhhB : WhhF;
    const float*  pre = dir ? pre_b : pre_f;
    __bf16*      hout = dir ? h_b  : h_f;

    v16bf Bf[4];                                  // resident W_hh fragments
#pragma unroll
    for (int kt = 0; kt < 4; ++kt)
        Bf[kt] = *(const v16bf*)(Whh + (size_t)ncol * HKP + kt * 32 + half * 16);

    for (int i = threadIdx.x; i < 16 * HKP; i += blockDim.x)
        (&hsh[0][0])[i] = f2bf(0.f);              // h0 = 0, pad cols stay 0
    __syncthreads();

    for (int step = 0; step < TLEN; ++step) {
        const int t = dir ? (TLEN - 1 - step) : step;
        // prefetch next step's pre rows while this step computes
        if (step + 1 < TLEN) {
            const int tn = dir ? (t - 1) : (t + 1);
#pragma unroll
            for (int r = 0; r < 8; ++r) {
                int m = r + half * 8;
                size_t tok = (size_t)(btile * 16 + m) * TLEN + tn;
                __builtin_prefetch(&pre[tok * HPAD + ncol], 0, 3);
            }
        }
        v8f acc = {};
        const int m = lane & 15;
#pragma unroll
        for (int kt = 0; kt < 4; ++kt) {
            const int ka = kt * 32 + half * 16;
            v8bf lo = *(const v8bf*)&hsh[m][ka];        // 16B ds loads
            v8bf hi = *(const v8bf*)&hsh[m][ka + 8];
            v16bf a;
#pragma unroll
            for (int e = 0; e < 8; ++e) { a[e] = lo[e]; a[e + 8] = hi[e]; }
            acc = __builtin_amdgcn_wmma_f32_16x16x32_bf16(false, a, false, Bf[kt],
                                                          (short)0, acc, false, false);
        }
        float hv[8];
#pragma unroll
        for (int r = 0; r < 8; ++r) {
            int mr = r + half * 8;
            size_t tok = (size_t)(btile * 16 + mr) * TLEN + t;
            hv[r] = tanhf(acc[r] + pre[tok * HPAD + ncol]);
        }
        __syncthreads();                          // all reads of hsh done
#pragma unroll
        for (int r = 0; r < 8; ++r) {
            int mr = r + half * 8;
            size_t tok = (size_t)(btile * 16 + mr) * TLEN + t;
            __bf16 hb16 = f2bf(hv[r]);
            hsh[mr][ncol] = hb16;
            hout[tok * HPAD + ncol] = hb16;
        }
        __syncthreads();
    }
}

// ---------------------------------------------------------------------------
// Stage 4: logits = [h_f|h_b] @ w_lin^T + b_lin  (M=131072, N=32, K=224)
// grid(2048, 2), block 128. Each 8-element A-run is aligned and never
// straddles the fwd/bwd boundary at k=112, so runs load as 16B vectors.
// ---------------------------------------------------------------------------
__global__ void k_emis_gemm(const __bf16* __restrict__ h_f, const __bf16* __restrict__ h_b,
                            const __bf16* __restrict__ Wlin, const float* __restrict__ blin,
                            float* __restrict__ logits) {
    const int lane  = threadIdx.x & 31;
    const int wave  = threadIdx.x >> 5;
    const int mtile = blockIdx.x * 4 + wave;
    const int nt    = blockIdx.y;                 // 0..1
    const int half  = lane >> 4, col = lane & 15;
    const size_t rowTok = (size_t)(mtile * 16 + col);
    const __bf16* wr = Wlin + (size_t)(nt * 16 + col) * KPAD;
    v8f acc = {};
#pragma unroll
    for (int kt = 0; kt < 7; ++kt) {
        const int ka = kt * 32 + half * 16;
        const int s0 = ka, s1 = ka + 8;
        const __bf16* p0 = (s0 < HPAD) ? (h_f + rowTok * HPAD + s0)
                                       : (h_b + rowTok * HPAD + (s0 - HPAD));
        const __bf16* p1 = (s1 < HPAD) ? (h_f + rowTok * HPAD + s1)
                                       : (h_b + rowTok * HPAD + (s1 - HPAD));
        v8bf lo = *(const v8bf*)p0;
        v8bf hi = *(const v8bf*)p1;
        v16bf a;
#pragma unroll
        for (int e = 0; e < 8; ++e) { a[e] = lo[e]; a[e + 8] = hi[e]; }
        v16bf bm = *(const v16bf*)(wr + kt * 32 + half * 16);
        acc = __builtin_amdgcn_wmma_f32_16x16x32_bf16(false, a, false, bm,
                                                      (short)0, acc, false, false);
    }
    const float bias = blin[nt * 16 + col];
#pragma unroll
    for (int r = 0; r < 8; ++r) {
        int m = r + half * 8;
        logits[(size_t)(mtile * 16 + m) * ESTR + nt * 16 + col] = acc[r] + bias;
    }
}

// ---------------------------------------------------------------------------
// Stage 5: in-place softmax over the 19 valid tag columns
// ---------------------------------------------------------------------------
__global__ void k_softmax(float* __restrict__ logits) {
    size_t tok = (size_t)blockIdx.x * blockDim.x + threadIdx.x;
    if (tok >= NTOK) return;
    float* p = logits + tok * ESTR;
    float m = p[0];
#pragma unroll
    for (int j = 1; j < KTAG; ++j) m = fmaxf(m, p[j]);
    float s = 0.f;
#pragma unroll
    for (int j = 0; j < KTAG; ++j) { float e = __expf(p[j] - m); p[j] = e; s += e; }
    float inv = 1.f / s;
#pragma unroll
    for (int j = 0; j < KTAG; ++j) p[j] *= inv;
}

// ---------------------------------------------------------------------------
// Stage 6: CRF log-likelihood per batch element (one wave32 per b, lane=tag)
// ---------------------------------------------------------------------------
__global__ void k_crf(const float* __restrict__ em, const int* __restrict__ y,
                      const float* __restrict__ start, const float* __restrict__ endv,
                      const float* __restrict__ trans, float* __restrict__ llh) {
    const int b = blockIdx.x, lane = threadIdx.x;
    const bool act = lane < KTAG;
    float tr[KTAG];
#pragma unroll
    for (int i = 0; i < KTAG; ++i) tr[i] = act ? trans[i * KTAG + lane] : 0.f;
    const float* emb = em + (size_t)b * TLEN * ESTR;
    const int*   yb  = y + (size_t)b * TLEN;
    float alpha = act ? (start[lane] + emb[lane]) : NEGBIG;

    for (int t = 1; t < TLEN; ++t) {
        const bool mb = (yb[t] != -1);
        float m = NEGBIG, s = 0.f;
#pragma unroll
        for (int i = 0; i < KTAG; ++i) {                 // online logsumexp
            float ai = __shfl(alpha, i, 32);
            float v  = ai + tr[i];
            if (v > m) { s = s * __expf(m - v) + 1.f; m = v; }
            else        s += __expf(v - m);
        }
        float nxt = m + __logf(s) + emb[(size_t)t * ESTR + lane];
        if (act && mb) alpha = nxt;
    }
    float v = act ? alpha + endv[lane] : NEGBIG;
    float m = v;
    for (int off = 16; off; off >>= 1) m = fmaxf(m, __shfl_xor(m, off, 32));
    float e = __expf(v - m);
    for (int off = 16; off; off >>= 1) e += __shfl_xor(e, off, 32);
    float logZ = m + __logf(e);

    if (lane == 0) {                                     // gold-path numerator
        int prev = (yb[0] == -1) ? 0 : yb[0];
        float num = start[prev] + emb[prev];
        for (int t = 1; t < TLEN; ++t) {
            int yt = yb[t];
            if (yt != -1) {
                num += trans[prev * KTAG + yt] + emb[(size_t)t * ESTR + yt];
                prev = yt;
            }
        }
        num += endv[prev];
        llh[b] = num - logZ;
    }
}

// ---------------------------------------------------------------------------
// Stage 7: loss = -sum_b llh[b]  (deterministic LDS tree reduction)
// ---------------------------------------------------------------------------
__global__ void k_loss(const float* __restrict__ llh, float* __restrict__ out) {
    __shared__ float sm[BSZ];
    int i = threadIdx.x;
    sm[i] = llh[i];
    __syncthreads();
    for (int s = BSZ / 2; s; s >>= 1) { if (i < s) sm[i] += sm[i + s]; __syncthreads(); }
    if (i == 0) out[0] = -sm[0];
}

// ---------------------------------------------------------------------------
// Stage 8: Viterbi (one wave32 per b, lane=tag; lane-0 backtrack)
// ---------------------------------------------------------------------------
__global__ void k_viterbi(const float* __restrict__ em, const int* __restrict__ y,
                          const float* __restrict__ start, const float* __restrict__ endv,
                          const float* __restrict__ trans, int* __restrict__ hist,
                          float* __restrict__ outlab) {
    const int b = blockIdx.x, lane = threadIdx.x;
    const bool act = lane < KTAG;
    float tr[KTAG];
#pragma unroll
    for (int i = 0; i < KTAG; ++i) tr[i] = act ? trans[i * KTAG + lane] : 0.f;
    const float* emb = em + (size_t)b * TLEN * ESTR;
    const int*   yb  = y + (size_t)b * TLEN;
    int* hb = hist + (size_t)b * (TLEN - 1) * KTAG;
    float delta = act ? (start[lane] + emb[lane]) : NEGBIG;

    for (int t = 1; t < TLEN; ++t) {
        const bool mb = (yb[t] != -1);
        float bmv = NEGBIG; int bi = 0;
#pragma unroll
        for (int i = 0; i < KTAG; ++i) {
            float ai = __shfl(delta, i, 32);
            float v  = ai + tr[i];
            if (v > bmv) { bmv = v; bi = i; }            // first max -> lowest idx
        }
        if (act) hb[(size_t)(t - 1) * KTAG + lane] = bi;
        float nxt = bmv + emb[(size_t)t * ESTR + lane];
        if (act && mb) delta = nxt;
    }
    float v = act ? delta + endv[lane] : NEGBIG;
    int idx = act ? lane : 31;
    for (int off = 16; off; off >>= 1) {
        float ov = __shfl_xor(v, off, 32);
        int   oi = __shfl_xor(idx, off, 32);
        if (ov > v || (ov == v && oi < idx)) { v = ov; idx = oi; }
    }
    if (lane == 0) {
        int tag = idx;
        float* ob = outlab + (size_t)b * TLEN;
        ob[TLEN - 1] = (yb[TLEN - 1] != -1) ? (float)tag : -1.f;
        for (int t = TLEN - 2; t >= 0; --t) {
            if (yb[t + 1] != -1) tag = hb[(size_t)t * KTAG + tag];
            ob[t] = (yb[t] != -1) ? (float)tag : -1.f;
        }
    }
}

// ---------------------------------------------------------------------------
extern "C" void kernel_launch(void* const* d_in, const int* in_sizes, int n_in,
                              void* d_out, int out_size, void* d_ws, size_t ws_size,
                              hipStream_t stream) {
    const float* x       = (const float*)d_in[0];
    const int*   y       = (const int*)  d_in[1];
    const float* w_ih_f  = (const float*)d_in[2];
    const float* w_hh_f  = (const float*)d_in[3];
    const float* b_ih_f  = (const float*)d_in[4];
    const float* b_hh_f  = (const float*)d_in[5];
    const float* w_ih_b  = (const float*)d_in[6];
    const float* w_hh_b  = (const float*)d_in[7];
    const float* b_ih_b  = (const float*)d_in[8];
    const float* b_hh_b  = (const float*)d_in[9];
    const float* w_lin   = (const float*)d_in[10];
    const float* b_lin   = (const float*)d_in[11];
    const float* startv  = (const float*)d_in[12];
    const float* endv    = (const float*)d_in[13];
    const float* trans   = (const float*)d_in[14];

    char* ws = (char*)d_ws;
    __bf16* Wcat  = (__bf16*)(ws + OFF_WCAT);
    __bf16* WhhF  = (__bf16*)(ws + OFF_WHHF);
    __bf16* WhhB  = (__bf16*)(ws + OFF_WHHB);
    __bf16* Wlin  = (__bf16*)(ws + OFF_WLIN);
    float*  bcat  = (float*) (ws + OFF_BCAT);
    float*  blin  = (float*) (ws + OFF_BLIN);
    float*  pre_f = (float*) (ws + OFF_PREF);
    float*  pre_b = (float*) (ws + OFF_PREB);
    __bf16* h_f   = (__bf16*)(ws + OFF_HF);
    __bf16* h_b   = (__bf16*)(ws + OFF_HB);
    float*  em    = (float*) (ws + OFF_EM);
    int*    hist  = (int*)   (ws + OFF_HIST);
    float*  llh   = (float*) (ws + OFF_LLH);

    const int prepN = KPAD * KPAD + 2 * HPAD * HKP + 32 * KPAD + KPAD + 32;
    k_prep<<<(prepN + 255) / 256, 256, 0, stream>>>(
        w_ih_f, w_hh_f, b_ih_f, b_hh_f, w_ih_b, w_hh_b, b_ih_b, b_hh_b,
        w_lin, b_lin, Wcat, WhhF, WhhB, Wlin, bcat, blin);

    k_input_gemm<<<dim3(NTOK / 64, 2), 128, 0, stream>>>(x, Wcat, bcat, pre_f, pre_b);

    k_rnn<<<dim3(BSZ / 16, 2), 224, 0, stream>>>(pre_f, pre_b, WhhF, WhhB, h_f, h_b);

    k_emis_gemm<<<dim3(NTOK / 64, 2), 128, 0, stream>>>(h_f, h_b, Wlin, blin, em);

    k_softmax<<<(NTOK + 255) / 256, 256, 0, stream>>>(em);

    k_crf<<<BSZ, 32, 0, stream>>>(em, y, startv, endv, trans, llh);

    k_loss<<<1, BSZ, 0, stream>>>(llh, (float*)d_out);

    k_viterbi<<<BSZ, 32, 0, stream>>>(em, y, startv, endv, trans, hist,
                                      (float*)d_out + 1);
}
